// MoEFFN_88527865905618
// MI455X (gfx1250) — matrix operand order repaired and checked by reading, not compiled, not run on hardware
//
#include <hip/hip_runtime.h>
#include <hip/hip_bf16.h>

// ---------------------------------------------------------------------------
// MoE FFN (top-2 of 8 experts), D=1024, H=4096, T=8192 tokens.
// Grouped bf16 WMMA GEMMs on gfx1250 (wave32, v_wmma_f32_16x16x32_bf16),
// tiles staged with GLOBAL_LOAD_ASYNC_TO_LDS_B128 (ASYNCcnt pipelining).
// Workspace usage: ~287 MB (bf16 x / fc1 / fc2 copies, bf16 h, buckets).
// ---------------------------------------------------------------------------

typedef __attribute__((ext_vector_type(16))) __bf16 v16bf;
typedef __attribute__((ext_vector_type(8)))  __bf16 v8bf;
typedef __attribute__((ext_vector_type(4)))  __bf16 v4bf;
typedef __attribute__((ext_vector_type(8)))  float  v8f;

#define MOE_D   1024
#define MOE_H   4096
#define MOE_E   8
#define MOE_T   8192   // B*S tokens
#define MOE_BM  128
#define MOE_BN  128
#define MOE_BK  32

__device__ __forceinline__ int imin(int a, int b) { return a < b ? a : b; }

__device__ __forceinline__ float gelu_tanh(float v) {
    float c = 0.7978845608028654f * (v + 0.044715f * v * v * v);
    return 0.5f * v * (1.0f + tanhf(c));
}

// Async DMA: copy 16B from per-lane global address into per-lane LDS offset.
// ISA §15.18.3 opcode 98, tracked by ASYNCcnt.
__device__ __forceinline__ void async_g2l_b128(unsigned lds_off, const void* gaddr) {
    asm volatile("global_load_async_to_lds_b128 %0, %1, off"
                 :: "v"(lds_off), "v"(gaddr) : "memory");
}
__device__ __forceinline__ void wait_asynccnt0() {
    asm volatile("s_wait_asynccnt 0x0" ::: "memory");
}

// ---------------------------------------------------------------------------
// Kernel 0: zero the output accumulator (fc2 scatters with atomicAdd) + counts
// ---------------------------------------------------------------------------
__global__ void moe_zero_kernel(float* __restrict__ out, size_t n,
                                int* __restrict__ counts) {
    size_t i = (size_t)blockIdx.x * blockDim.x + threadIdx.x;
    if (i < MOE_E) counts[i] = 0;
    size_t stride = (size_t)gridDim.x * blockDim.x;
    float4* o4 = (float4*)out;
    for (size_t j = i; j < n / 4; j += stride)
        o4[j] = make_float4(0.f, 0.f, 0.f, 0.f);
}

// ---------------------------------------------------------------------------
// Kernel 1: fp32 -> bf16 conversion (x, fc1_w, fc2_w), vectorized x4
// ---------------------------------------------------------------------------
__global__ void moe_cvt_bf16_kernel(const float* __restrict__ src,
                                    __bf16* __restrict__ dst, size_t n4) {
    size_t i = (size_t)blockIdx.x * blockDim.x + threadIdx.x;
    size_t stride = (size_t)gridDim.x * blockDim.x;
    const float4* s4 = (const float4*)src;
    v4bf* d4 = (v4bf*)dst;
    for (size_t j = i; j < n4; j += stride) {
        float4 v = s4[j];
        v4bf o = { (__bf16)v.x, (__bf16)v.y, (__bf16)v.z, (__bf16)v.w };
        d4[j] = o;
    }
}

// ---------------------------------------------------------------------------
// Kernel 2: router — one wave32 per token. logits = (x+se)·W_r^T + b,
// top-2, softmax over the pair, scatter token into per-expert buckets.
// ---------------------------------------------------------------------------
__global__ __launch_bounds__(256)
void moe_router_kernel(const float* __restrict__ x,
                       const float* __restrict__ semb,
                       const int* __restrict__ scale_idx,
                       const float* __restrict__ rw,
                       const float* __restrict__ rb,
                       int* __restrict__ counts,
                       int* __restrict__ bucket_tok,
                       float* __restrict__ bucket_gate) {
    int wave = threadIdx.x >> 5;
    int lane = threadIdx.x & 31;
    int tok = blockIdx.x * 8 + wave;
    if (tok >= MOE_T) return;

    const float* xr = x + (size_t)tok * MOE_D;
    const float* se = semb + (size_t)scale_idx[0] * MOE_D;

    float xs[MOE_D / 32];
    #pragma unroll
    for (int i = 0; i < MOE_D / 32; ++i)
        xs[i] = xr[lane + 32 * i] + se[lane + 32 * i];

    float logits[MOE_E];
    #pragma unroll
    for (int e = 0; e < MOE_E; ++e) {
        const float* w = rw + (size_t)e * MOE_D;
        float p = 0.f;
        #pragma unroll
        for (int i = 0; i < MOE_D / 32; ++i)
            p += xs[i] * w[lane + 32 * i];
        // wave32 butterfly reduction
        #pragma unroll
        for (int m = 16; m > 0; m >>= 1)
            p += __shfl_xor(p, m, 32);
        logits[e] = p + rb[e];
    }

    // top-2 (ties -> first index, matches lax.top_k)
    int e0 = 0;
    #pragma unroll
    for (int e = 1; e < MOE_E; ++e)
        if (logits[e] > logits[e0]) e0 = e;
    int e1 = (e0 == 0) ? 1 : 0;
    #pragma unroll
    for (int e = 0; e < MOE_E; ++e)
        if (e != e0 && logits[e] > logits[e1]) e1 = e;

    float p1 = __expf(logits[e1] - logits[e0]);   // p0 = 1
    float inv = 1.0f / (1.0f + p1);
    float w0 = inv, w1 = p1 * inv;

    if (lane == 0) {
        int pos = atomicAdd(&counts[e0], 1);
        bucket_tok[e0 * MOE_T + pos] = tok;
        bucket_gate[e0 * MOE_T + pos] = w0;
        pos = atomicAdd(&counts[e1], 1);
        bucket_tok[e1 * MOE_T + pos] = tok;
        bucket_gate[e1 * MOE_T + pos] = w1;
    }
}

// ---------------------------------------------------------------------------
// Kernel 3: 8-element exclusive scan (compacts h rows across experts)
// ---------------------------------------------------------------------------
__global__ void moe_scan_kernel(const int* __restrict__ counts,
                                int* __restrict__ offsets) {
    if (threadIdx.x == 0) {
        int a = 0;
        for (int e = 0; e < MOE_E; ++e) { offsets[e] = a; a += counts[e]; }
    }
}

// ---------------------------------------------------------------------------
// Grouped GEMM. FC1: h = gelu(gather(x) @ W1^T + b1)  (M=cnt, N=H, K=D)
//               FC2: out += gate * (h @ W2^T + b2)     (M=cnt, N=D, K=H)
// Both are NT GEMMs: A rows and B(=W) rows are K-contiguous bf16.
// 256 threads = 8 waves; wave grid 4(M) x 2(N); 8 accum tiles / wave;
// 128x128x32 block tile, double-buffered LDS filled by async-to-LDS DMA.
// ---------------------------------------------------------------------------
union FragBF { v16bf v; v8bf h[2]; };

template<bool FC1>
__global__ __launch_bounds__(256)
void moe_gemm_kernel(const __bf16* __restrict__ Abase,   // xb (FC1) / h (FC2)
                     const __bf16* __restrict__ Wb,      // bf16 weights, [E][N][K]
                     const float*  __restrict__ bias,    // [E][N]
                     __bf16* __restrict__ h_out,         // FC1 output
                     float*  __restrict__ out,           // FC2 output
                     const int* __restrict__ counts,
                     const int* __restrict__ offsets,
                     const int* __restrict__ bucket_tok,
                     const float* __restrict__ bucket_gate,
                     int N, int K) {
    __shared__ alignas(16) __bf16 sA[2][MOE_BM][MOE_BK];
    __shared__ alignas(16) __bf16 sB[2][MOE_BN][MOE_BK];
    __shared__ int    sTok[MOE_BM];
    __shared__ float  sGate[MOE_BM];

    const int e   = blockIdx.z;
    const int cnt = counts[e];
    const int m0  = blockIdx.y * MOE_BM;
    if (m0 >= cnt) return;
    const int off = offsets[e];
    const int n0  = blockIdx.x * MOE_BN;
    const int t   = threadIdx.x;

    if (t < MOE_BM) {
        int slot = imin(m0 + t, cnt - 1);
        sTok[t]  = bucket_tok[e * MOE_T + slot];
        sGate[t] = bucket_gate[e * MOE_T + slot];
    }
    __syncthreads();

    // --- cooperative tile staging: 256 threads, 32B per tile each, all via
    //     global_load_async_to_lds_b128 (per-lane gather is free) ---
    const int lrow = t >> 1;            // 0..127
    const int lseg = (t & 1) * 16;      // 0 or 16 (bf16 elements)
    size_t arow = FC1 ? (size_t)sTok[lrow]
                      : (size_t)(off + imin(m0 + lrow, cnt - 1));
    const __bf16* aSrc = Abase + arow * (size_t)K + lseg;
    const __bf16* bSrc = Wb + ((size_t)e * N + (n0 + lrow)) * (size_t)K + lseg;

    const unsigned ldsA = (unsigned)(uintptr_t)&sA[0][lrow][lseg];
    const unsigned ldsB = (unsigned)(uintptr_t)&sB[0][lrow][lseg];
    const unsigned bufAstride = sizeof(sA[0]);   // 8 KB
    const unsigned bufBstride = sizeof(sB[0]);   // 8 KB

    auto stage_tile = [&](int buf, int kt) {
        const __bf16* ap = aSrc + (size_t)kt * MOE_BK;
        const __bf16* bp = bSrc + (size_t)kt * MOE_BK;
        const unsigned ao = ldsA + (unsigned)buf * bufAstride;
        const unsigned bo = ldsB + (unsigned)buf * bufBstride;
        async_g2l_b128(ao,      ap);
        async_g2l_b128(ao + 16, ap + 8);
        async_g2l_b128(bo,      bp);
        async_g2l_b128(bo + 16, bp + 8);
    };

    // --- wave tiling ---
    const int lane  = t & 31;
    const int wave  = t >> 5;
    const int wm    = (wave >> 1) * 32;  // 4 waves along M (2 m-tiles each)
    const int wn    = (wave & 1) * 64;   // 2 waves along N (4 n-tiles each)
    const int lm    = lane & 15;
    const int lhalf = lane >> 4;

    v8f acc[2][4] = {};

    // prologue: DMA tile 0 into buffer 0
    stage_tile(0, 0);
    wait_asynccnt0();
    __syncthreads();

    const int nk = K / MOE_BK;
    for (int kt = 0; kt < nk; ++kt) {
        const int cur = kt & 1;
        const bool nx = (kt + 1) < nk;
        if (nx) {
            stage_tile(cur ^ 1, kt + 1);               // DMA overlaps WMMA below
            __builtin_prefetch(bSrc + (size_t)(kt + 2) * MOE_BK, 0, 1);
        }

        // Fragment loads per ISA §7.12.2 layouts (all ds_load_b128).
        FragBF fa[2], fb[4];
        #pragma unroll
        for (int i = 0; i < 2; ++i) {
            const int r = wm + i * 16 + lm;
            fa[i].h[0] = *(const v8bf*)&sA[cur][r][lhalf * 8];
            fa[i].h[1] = *(const v8bf*)&sA[cur][r][lhalf * 8 + 16];
        }
        #pragma unroll
        for (int j = 0; j < 4; ++j) {
            const int r = wn + j * 16 + lm;
            fb[j].h[0] = *(const v8bf*)&sB[cur][r][lhalf * 8];
            fb[j].h[1] = *(const v8bf*)&sB[cur][r][lhalf * 8 + 16];
        }
        #pragma unroll
        for (int i = 0; i < 2; ++i)
            #pragma unroll
            for (int j = 0; j < 4; ++j)
                acc[i][j] = __builtin_amdgcn_wmma_f32_16x16x32_bf16(
                    false, fa[i].v, false, fb[j].v,
                    (short)0, acc[i][j], false, false);

        // DMA into next buffer must be LDS-visible before anyone passes the
        // barrier: per-wave ASYNCcnt drain, then workgroup barrier.
        if (nx) wait_asynccnt0();
        __syncthreads();
    }

    // --- epilogue. C layout: VGPR j -> row j + 8*lhalf, col = lm ---
    // bias depends only on the column -> hoist out of the row loops
    float bn[4];
    #pragma unroll
    for (int nt = 0; nt < 4; ++nt)
        bn[nt] = bias[(size_t)e * N + (n0 + wn + nt * 16 + lm)];

    #pragma unroll
    for (int i = 0; i < 2; ++i) {
        #pragma unroll
        for (int j = 0; j < 8; ++j) {
            const int lrowc = wm + i * 16 + lhalf * 8 + j;   // row in block
            const int slot  = m0 + lrowc;
            if (slot >= cnt) continue;
            if (FC1) {
                __bf16* hrow = h_out + (size_t)(off + slot) * N;
                #pragma unroll
                for (int nt = 0; nt < 4; ++nt) {
                    const int n = n0 + wn + nt * 16 + lm;
                    float v = acc[i][nt][j] + bn[nt];
                    hrow[n] = (__bf16)gelu_tanh(v);
                }
            } else {
                const int tokr = sTok[lrowc];
                const float g  = sGate[lrowc];
                float* orow = out + (size_t)tokr * N;
                #pragma unroll
                for (int nt = 0; nt < 4; ++nt) {
                    const int n = n0 + wn + nt * 16 + lm;
                    float v = g * (acc[i][nt][j] + bn[nt]);
                    atomicAdd(&orow[n], v);
                }
            }
        }
    }
}

// ---------------------------------------------------------------------------
extern "C" void kernel_launch(void* const* d_in, const int* in_sizes, int n_in,
                              void* d_out, int out_size, void* d_ws, size_t ws_size,
                              hipStream_t stream) {
    const float* x       = (const float*)d_in[0];
    const float* semb    = (const float*)d_in[1];
    const float* rw      = (const float*)d_in[2];
    const float* rb      = (const float*)d_in[3];
    const float* fc1_w   = (const float*)d_in[4];
    const float* fc1_b   = (const float*)d_in[5];
    const float* fc2_w   = (const float*)d_in[6];
    const float* fc2_b   = (const float*)d_in[7];
    const int*   sidx    = (const int*)d_in[8];
    float* out = (float*)d_out;

    // workspace carve (256B aligned)
    char* p = (char*)d_ws;
    auto carve = [&](size_t bytes) {
        void* r = (void*)p;
        p += (bytes + 255) & ~(size_t)255;
        return r;
    };
    __bf16* xb   = (__bf16*)carve((size_t)MOE_T * MOE_D * 2);
    __bf16* w1b  = (__bf16*)carve((size_t)MOE_E * MOE_H * MOE_D * 2);
    __bf16* w2b  = (__bf16*)carve((size_t)MOE_E * MOE_D * MOE_H * 2);
    __bf16* hbuf = (__bf16*)carve((size_t)MOE_T * 2 * MOE_H * 2);
    int*   counts  = (int*)carve(MOE_E * sizeof(int));
    int*   offsets = (int*)carve(MOE_E * sizeof(int));
    int*   btok    = (int*)carve((size_t)MOE_E * MOE_T * sizeof(int));
    float* bgate   = (float*)carve((size_t)MOE_E * MOE_T * sizeof(float));

    // 0) zero out + counts
    moe_zero_kernel<<<2048, 256, 0, stream>>>(out, (size_t)MOE_T * MOE_D, counts);

    // 1) fp32 -> bf16
    moe_cvt_bf16_kernel<<<2048, 256, 0, stream>>>(x, xb, (size_t)MOE_T * MOE_D / 4);
    moe_cvt_bf16_kernel<<<8192, 256, 0, stream>>>(fc1_w, w1b, (size_t)MOE_E * MOE_H * MOE_D / 4);
    moe_cvt_bf16_kernel<<<8192, 256, 0, stream>>>(fc2_w, w2b, (size_t)MOE_E * MOE_D * MOE_H / 4);

    // 2) routing + bucket scatter
    moe_router_kernel<<<MOE_T / 8, 256, 0, stream>>>(x, semb, sidx, rw, rb,
                                                     counts, btok, bgate);
    // 3) prefix scan
    moe_scan_kernel<<<1, 32, 0, stream>>>(counts, offsets);

    // 4) fc1 + gelu -> h (bf16)
    dim3 g1(MOE_H / MOE_BN, MOE_T / MOE_BM, MOE_E);
    moe_gemm_kernel<true><<<g1, 256, 0, stream>>>(
        xb, w1b, fc1_b, hbuf, nullptr,
        counts, offsets, btok, bgate, MOE_H, MOE_D);

    // 5) fc2 + bias, gate-weighted scatter-add into out
    dim3 g2(MOE_D / MOE_BN, MOE_T / MOE_BM, MOE_E);
    moe_gemm_kernel<false><<<g2, 256, 0, stream>>>(
        hbuf, w2b, fc2_b, nullptr, out,
        counts, offsets, btok, bgate, MOE_D, MOE_H);

    (void)in_sizes; (void)n_in; (void)out_size; (void)ws_size;
}
